// HANLayer_9113920602498
// MI455X (gfx1250) — compile-verified, matrix operand order checked by reading, not run on hardware
//
#include <hip/hip_runtime.h>
#include <hip/hip_bf16.h>

typedef __attribute__((ext_vector_type(16))) _Float16 v16h;
typedef __attribute__((ext_vector_type(8)))  float    v8f;

#define NN 50000
#define EE 600000
#define MM 3
#define NEG_SLOPE 0.2f
#define TPW 5   // row tiles per wave: 3125 = 5*625, 9375 = 5*1875 (no remainder)

// ---------------------------------------------------------------------------
// float atomic max via int/uint ordering trick (works for mixed signs given
// init = -3e38): positive floats order as positive ints (atomicMax on int),
// negative floats order inversely as unsigned (atomicMin on uint).
// ---------------------------------------------------------------------------
__device__ __forceinline__ void atomicMaxFloat(float* addr, float val) {
    if (val >= 0.0f) atomicMax((int*)addr, __float_as_int(val));
    else             atomicMin((unsigned int*)addr, __float_as_uint(val));
}

// ---------------------------------------------------------------------------
// init per-metapath scratch: emax=-3e38, denom=0, zm(acc)=0
// ---------------------------------------------------------------------------
__global__ void init_kernel(float* __restrict__ emax, float* __restrict__ denom,
                            float* __restrict__ zm) {
    int i = blockIdx.x * blockDim.x + threadIdx.x;
    if (i < NN * 4) { emax[i] = -3.0e38f; denom[i] = 0.0f; }
    if (i < NN * 128) zm[i] = 0.0f;
}

__global__ void zero_small_kernel(float* __restrict__ p, int n) {
    int i = threadIdx.x;
    if (i < n) p[i] = 0.0f;
}

// ---------------------------------------------------------------------------
// WMMA GEMM: h[nrows,128] = A[nrows,128] @ B[128,128]
// block = 256 threads (8 waves); wave w owns col tile w (16 cols);
// each wave processes TPW=5 row tiles (80 rows/block): B fragment loaded once
// per K-step and reused across 5 WMMAs; K loop fully unrolled -> 20 WMMAs.
// No guards anywhere: tile counts divide exactly, EXEC is always all-ones.
// A frag (16x32 f16): lane<16 -> K {0..7,16..23}; lane>=16 -> K {8..15,24..31}
// B frag mirrored with col = lane&15.
// ---------------------------------------------------------------------------
__global__ void gemm_xw_wmma(const float* __restrict__ x,
                             const float* __restrict__ Wm,
                             float* __restrict__ h) {
    const int wave  = threadIdx.x >> 5;
    const int lane  = threadIdx.x & 31;
    const int tile0 = blockIdx.x * TPW;
    const int col   = wave * 16 + (lane & 15);
    const int kOff  = (lane >> 4) * 8;
    const int rl    = lane & 15;

    v8f c[TPW] = {{}, {}, {}, {}, {}};
#pragma unroll
    for (int k0 = 0; k0 < 128; k0 += 32) {
        v16h b;
#pragma unroll
        for (int j = 0; j < 8; ++j) {
            b[j]     = (_Float16)Wm[(k0 + kOff + j) * 128 + col];
            b[j + 8] = (_Float16)Wm[(k0 + 16 + kOff + j) * 128 + col];
        }
#pragma unroll
        for (int t = 0; t < TPW; ++t) {
            const size_t row = (size_t)(tile0 + t) * 16 + rl;
            v16h a;
#pragma unroll
            for (int j = 0; j < 8; ++j) {
                a[j]     = (_Float16)x[row * 128 + k0 + kOff + j];
                a[j + 8] = (_Float16)x[row * 128 + k0 + 16 + kOff + j];
            }
            c[t] = __builtin_amdgcn_wmma_f32_16x16x32_f16(false, a, false, b,
                                                          (short)0, c[t],
                                                          false, false);
        }
    }
    // C layout: element r -> local row r + (lane<16 ? 0 : 8), col as above
    const int rbase = (lane >> 4) << 3;
#pragma unroll
    for (int t = 0; t < TPW; ++t) {
        const size_t rowb = (size_t)(tile0 + t) * 16 + rbase;
#pragma unroll
        for (int r = 0; r < 8; ++r)
            h[(rowb + r) * 128 + col] = c[t][r];
    }
}

// ---------------------------------------------------------------------------
// alpha_s[n,h] = sum_d h[n,h,d]*att_src[h,d] ; same for alpha_d.
// block = 128 threads = 4 waves; wave w == head w; wave32 shuffle reduce.
// ---------------------------------------------------------------------------
__global__ void alpha_kernel(const float* __restrict__ h,
                             const float* __restrict__ a_src,
                             const float* __restrict__ a_dst,
                             float* __restrict__ alpha_s,
                             float* __restrict__ alpha_d) {
    const int n    = blockIdx.x;
    const int t    = threadIdx.x;       // 0..127 -> (head, d)
    const int head = t >> 5;
    const int lane = t & 31;
    float hv = h[(size_t)n * 128 + t];
    float ps = hv * a_src[t];
    float pd = hv * a_dst[t];
#pragma unroll
    for (int off = 16; off; off >>= 1) {
        ps += __shfl_down(ps, off, 32);
        pd += __shfl_down(pd, off, 32);
    }
    if (lane == 0) {
        alpha_s[n * 4 + head] = ps;
        alpha_d[n * 4 + head] = pd;
    }
}

// ---------------------------------------------------------------------------
// edge pass A: segment max of leaky_relu(as[src]+ad[dst]) per (dst, head)
// one thread per edge (self-loops appended for e >= E)
// ---------------------------------------------------------------------------
__global__ void edge_max_kernel(const int* __restrict__ srcI,
                                const int* __restrict__ dstI,
                                const float* __restrict__ alpha_s,
                                const float* __restrict__ alpha_d,
                                float* __restrict__ emax) {
    int e = blockIdx.x * blockDim.x + threadIdx.x;
    const int total = EE + NN;
    if (e >= total) return;
    int s, d;
    if (e < EE) { s = srcI[e]; d = dstI[e]; }
    else        { s = d = e - EE; }
#pragma unroll
    for (int hh = 0; hh < 4; ++hh) {
        float v = alpha_s[s * 4 + hh] + alpha_d[d * 4 + hh];
        v = v > 0.0f ? v : NEG_SLOPE * v;
        atomicMaxFloat(&emax[d * 4 + hh], v);
    }
}

// ---------------------------------------------------------------------------
// edge pass B: e = exp(lrelu(..) - emax[dst]); denom[dst,h] += e;
// acc[dst, :] += e * h[src, :]   (unnormalized; divide in post pass)
// one wave per edge; lane L handles dims 4L..4L+3 (all in head L>>3)
// ---------------------------------------------------------------------------
__global__ void edge_msg_kernel(const int* __restrict__ srcI,
                                const int* __restrict__ dstI,
                                const float* __restrict__ alpha_s,
                                const float* __restrict__ alpha_d,
                                const float* __restrict__ emax,
                                const float* __restrict__ hbuf,
                                float* __restrict__ denom,
                                float* __restrict__ acc) {
    int gid  = blockIdx.x * blockDim.x + threadIdx.x;
    int e    = gid >> 5;
    int lane = gid & 31;
    const int total = EE + NN;
    if (e >= total) return;
    int s, d;
    if (e < EE) { s = srcI[e]; d = dstI[e]; }
    else        { s = d = e - EE; }
    const int head = lane >> 3;
    float v = alpha_s[s * 4 + head] + alpha_d[d * 4 + head];
    v = v > 0.0f ? v : NEG_SLOPE * v;
    float ee = __expf(v - emax[d * 4 + head]);
    if ((lane & 7) == 0) atomicAdd(&denom[d * 4 + head], ee);
    const float4 hv = *(const float4*)(hbuf + (size_t)s * 128 + lane * 4);
    float* ap = acc + (size_t)d * 128 + lane * 4;
    atomicAdd(ap + 0, ee * hv.x);
    atomicAdd(ap + 1, ee * hv.y);
    atomicAdd(ap + 2, ee * hv.z);
    atomicAdd(ap + 3, ee * hv.w);
}

// ---------------------------------------------------------------------------
// per-node epilogue (in place on zm): y = acc/denom + bias; LayerNorm; ELU
// block = 128 threads (4 waves); cross-wave reduce through LDS
// ---------------------------------------------------------------------------
__global__ void post_kernel(float* __restrict__ zm,
                            const float* __restrict__ denom,
                            const float* __restrict__ bias_m,
                            const float* __restrict__ ln_gamma,
                            const float* __restrict__ ln_beta) {
    __shared__ float red[8];
    const int n    = blockIdx.x;
    const int t    = threadIdx.x;
    const int head = t >> 5;
    const int lane = t & 31;
    float y = zm[(size_t)n * 128 + t] / (denom[n * 4 + head] + 1e-16f) + bias_m[t];

    float s = y;
#pragma unroll
    for (int off = 16; off; off >>= 1) s += __shfl_down(s, off, 32);
    if (lane == 0) red[head] = s;
    __syncthreads();
    float mean = (red[0] + red[1] + red[2] + red[3]) * (1.0f / 128.0f);

    float dy = y - mean;
    float s2 = dy * dy;
#pragma unroll
    for (int off = 16; off; off >>= 1) s2 += __shfl_down(s2, off, 32);
    if (lane == 0) red[4 + head] = s2;
    __syncthreads();
    float var = (red[4] + red[5] + red[6] + red[7]) * (1.0f / 128.0f);

    float v = dy * rsqrtf(var + 1e-5f) * ln_gamma[t] + ln_beta[t];
    v = v > 0.0f ? v : (__expf(v) - 1.0f);   // ELU
    zm[(size_t)n * 128 + t] = v;
}

// ---------------------------------------------------------------------------
// semantic attention scores via WMMA:
// for each row r of z[M*N,128]: s_r = tanh(z_r @ W1 + b1) . W2 ; wsum[m] += s_r
// block = 256 threads (8 waves x 16-col tiles), TPW=5 row tiles (80 rows/block)
// ---------------------------------------------------------------------------
__global__ void semantic_wmma(const float* __restrict__ z,
                              const float* __restrict__ W1,
                              const float* __restrict__ b1,
                              const float* __restrict__ W2,
                              float* __restrict__ wsum) {
    __shared__ float rowsum[16 * TPW];
    const int wave  = threadIdx.x >> 5;
    const int lane  = threadIdx.x & 31;
    const int tile0 = blockIdx.x * TPW;
    const int col   = wave * 16 + (lane & 15);
    const int kOff  = (lane >> 4) * 8;
    const int rl    = lane & 15;

    if (threadIdx.x < 16 * TPW) rowsum[threadIdx.x] = 0.0f;
    __syncthreads();

    v8f c[TPW] = {{}, {}, {}, {}, {}};
#pragma unroll
    for (int k0 = 0; k0 < 128; k0 += 32) {
        v16h b;
#pragma unroll
        for (int j = 0; j < 8; ++j) {
            b[j]     = (_Float16)W1[(k0 + kOff + j) * 128 + col];
            b[j + 8] = (_Float16)W1[(k0 + 16 + kOff + j) * 128 + col];
        }
#pragma unroll
        for (int t = 0; t < TPW; ++t) {
            const size_t row = (size_t)(tile0 + t) * 16 + rl;
            v16h a;
#pragma unroll
            for (int j = 0; j < 8; ++j) {
                a[j]     = (_Float16)z[row * 128 + k0 + kOff + j];
                a[j + 8] = (_Float16)z[row * 128 + k0 + 16 + kOff + j];
            }
            c[t] = __builtin_amdgcn_wmma_f32_16x16x32_f16(false, a, false, b,
                                                          (short)0, c[t],
                                                          false, false);
        }
    }
    const float bc = b1[col];
    const float w2 = W2[col];
    const int rloc = (lane >> 4) << 3;
#pragma unroll
    for (int t = 0; t < TPW; ++t) {
#pragma unroll
        for (int r = 0; r < 8; ++r) {
            float v = tanhf(c[t][r] + bc) * w2;
            atomicAdd(&rowsum[t * 16 + rloc + r], v);
        }
    }
    __syncthreads();
    if (threadIdx.x < 16 * TPW) {
        int rowg = tile0 * 16 + threadIdx.x;   // always < M*N (exact division)
        atomicAdd(&wsum[rowg / NN], rowsum[threadIdx.x]);
    }
}

// ---------------------------------------------------------------------------
// beta = softmax(wsum / N) over M; write att_mp to output tail
// ---------------------------------------------------------------------------
__global__ void beta_kernel(const float* __restrict__ wsum,
                            float* __restrict__ beta_out,
                            float* __restrict__ att_out) {
    float w0 = wsum[0] * (1.0f / NN);
    float w1 = wsum[1] * (1.0f / NN);
    float w2 = wsum[2] * (1.0f / NN);
    float mx = fmaxf(w0, fmaxf(w1, w2));
    float e0 = __expf(w0 - mx), e1 = __expf(w1 - mx), e2 = __expf(w2 - mx);
    float inv = 1.0f / (e0 + e1 + e2);
    beta_out[0] = e0 * inv; beta_out[1] = e1 * inv; beta_out[2] = e2 * inv;
    att_out[0]  = e0 * inv; att_out[1]  = e1 * inv; att_out[2]  = e2 * inv;
}

// ---------------------------------------------------------------------------
// out_emb[n,:] = sum_m beta[m] * z[m,n,:]
// ---------------------------------------------------------------------------
__global__ void combine_kernel(const float* __restrict__ z,
                               const float* __restrict__ beta_w,
                               float* __restrict__ out) {
    size_t i = (size_t)blockIdx.x * blockDim.x + threadIdx.x;
    if (i >= (size_t)NN * 128) return;
    const size_t stride = (size_t)NN * 128;
    out[i] = beta_w[0] * z[i] + beta_w[1] * z[stride + i] + beta_w[2] * z[2 * stride + i];
}

// ---------------------------------------------------------------------------
extern "C" void kernel_launch(void* const* d_in, const int* in_sizes, int n_in,
                              void* d_out, int out_size, void* d_ws, size_t ws_size,
                              hipStream_t stream) {
    const float* x       = (const float*)d_in[0];
    const int*   mp      = (const int*)d_in[1];      // [M,2,E] indices
    const float* W       = (const float*)d_in[2];    // [M,128,128]
    const float* att_src = (const float*)d_in[3];    // [M,4,32]
    const float* att_dst = (const float*)d_in[4];
    const float* bias    = (const float*)d_in[5];    // [M,128]
    const float* gamma   = (const float*)d_in[6];    // [128]
    const float* betaLN  = (const float*)d_in[7];    // [128]
    const float* semW1   = (const float*)d_in[8];    // [128,128]
    const float* semb1   = (const float*)d_in[9];    // [128]
    const float* semW2   = (const float*)d_in[10];   // [128]
    float* out = (float*)d_out;                      // [N*128] ++ [M]

    float* ws      = (float*)d_ws;
    float* z       = ws;                               // M*N*128
    float* hbuf    = z + (size_t)MM * NN * 128;        // N*128
    float* alpha_s = hbuf + (size_t)NN * 128;          // N*4
    float* alpha_d = alpha_s + NN * 4;                 // N*4
    float* emax    = alpha_d + NN * 4;                 // N*4
    float* denom   = emax + NN * 4;                    // N*4
    float* wsum    = denom + NN * 4;                   // 4
    float* betaBuf = wsum + 4;                         // 4

    zero_small_kernel<<<1, 32, 0, stream>>>(wsum, 4);

    const int total_edges = EE + NN;
    const int gemm_blocks = (NN / 16) / TPW;           // 3125/5 = 625
    const int sem_blocks  = ((MM * NN) / 16) / TPW;    // 9375/5 = 1875

    for (int m = 0; m < MM; ++m) {
        float* zm = z + (size_t)m * NN * 128;
        const int* srcI = mp + (size_t)m * 2 * EE;
        const int* dstI = srcI + EE;

        init_kernel<<<(NN * 128 + 255) / 256, 256, 0, stream>>>(emax, denom, zm);
        gemm_xw_wmma<<<gemm_blocks, 256, 0, stream>>>(
            x, W + (size_t)m * 128 * 128, hbuf);
        alpha_kernel<<<NN, 128, 0, stream>>>(hbuf, att_src + m * 128, att_dst + m * 128,
                                             alpha_s, alpha_d);
        edge_max_kernel<<<(total_edges + 255) / 256, 256, 0, stream>>>(
            srcI, dstI, alpha_s, alpha_d, emax);
        edge_msg_kernel<<<(total_edges * 32 + 255) / 256, 256, 0, stream>>>(
            srcI, dstI, alpha_s, alpha_d, emax, hbuf, denom, zm);
        post_kernel<<<NN, 128, 0, stream>>>(zm, denom, bias + m * 128, gamma, betaLN);
    }

    semantic_wmma<<<sem_blocks, 256, 0, stream>>>(z, semW1, semb1, semW2, wsum);
    beta_kernel<<<1, 1, 0, stream>>>(wsum, betaBuf, out + (size_t)NN * 128);
    combine_kernel<<<(int)(((size_t)NN * 128 + 255) / 256), 256, 0, stream>>>(
        z, betaBuf, out);
}
